// MultiHeadGraphAtten_5677946765825
// MI455X (gfx1250) — matrix operand order repaired and checked by reading, not compiled, not run on hardware
//
#include <hip/hip_runtime.h>

typedef __attribute__((ext_vector_type(16))) _Float16 v16h;
typedef __attribute__((ext_vector_type(8)))  float    v8f;
typedef __attribute__((ext_vector_type(4)))  int      v4i;

#define NEG_SLOPE 0.01f

static constexpr int Bd = 4, Nd = 2048, INd = 256, Fd = 64, Hd = 8;

// ---------------------------------------------------------------------------
// Async global->LDS staging (GLOBAL_LOAD_ASYNC_TO_LDS_B128, ASYNCcnt) with
// sync fallback if the builtin is not declared on this toolchain.
// The builtin's params are v4i* in AS1 (global) / AS3 (LDS).
// ---------------------------------------------------------------------------
#if defined(__has_builtin)
#if __has_builtin(__builtin_amdgcn_global_load_async_to_lds_b128)
#define ASYNC_LDS 1
#endif
#endif
#ifndef ASYNC_LDS
#define ASYNC_LDS 0
#endif

typedef __attribute__((address_space(1))) v4i gas_v4i;
typedef __attribute__((address_space(3))) v4i las_v4i;

__device__ __forceinline__ void copy16_g2l(const _Float16* g, _Float16* l) {
#if ASYNC_LDS
  __builtin_amdgcn_global_load_async_to_lds_b128(
      (gas_v4i*)g, (las_v4i*)l, 0, 0);
#else
  *(float4*)l = *(const float4*)g;
#endif
}

__device__ __forceinline__ void wait_async() {
#if ASYNC_LDS
#if __has_builtin(__builtin_amdgcn_s_wait_asynccnt)
  __builtin_amdgcn_s_wait_asynccnt(0);
#else
  asm volatile("s_wait_asynccnt 0x0" ::: "memory");
#endif
#endif
}

// ---------------------------------------------------------------------------
// Kernel 0: f32 -> f16 convert
// ---------------------------------------------------------------------------
__global__ void k_f32_to_f16(const float* __restrict__ in, _Float16* __restrict__ out, int n) {
  int i = blockIdx.x * blockDim.x + threadIdx.x;
  if (i < n) out[i] = (_Float16)in[i];
}

// ---------------------------------------------------------------------------
// Kernel 1: V = X @ vW + vb   (M = B*N = 8192, K = 256, N = F = 64)
// 64x64 output tile per block (256 threads = 8 waves), K-step 32,
// WMMA f16->f32, double-buffered async LDS staging.
// ---------------------------------------------------------------------------
#define XS_STRIDE 40   // 32 + 8 halves pad (80B, 16B aligned)
#define WS_STRIDE 72   // 64 + 8 halves pad (144B, 16B aligned)

__global__ __launch_bounds__(256) void k_gemm_v(
    const _Float16* __restrict__ Xh, const _Float16* __restrict__ Wh,
    const float* __restrict__ vb,
    float* __restrict__ Vf, _Float16* __restrict__ Vh) {
  __shared__ _Float16 Xs[2][64 * XS_STRIDE];
  __shared__ _Float16 Ws[2][32 * WS_STRIDE];

  const int t    = threadIdx.x;
  const int lane = t & 31;
  const int wv   = t >> 5;
  const int isub  = wv >> 1;        // 0..3 : 16-row subtile
  const int fhalf = wv & 1;         // 0..1 : 32-col half
  const int m0   = blockIdx.x * 64;
  const int mrow  = lane & 15;
  const int khalf = lane >> 4;

  auto load_tiles = [&](int k0, int buf) {
    int idx = t * 8;
    {
      int r = idx >> 5, c = idx & 31;
      copy16_g2l(Xh + (size_t)(m0 + r) * INd + k0 + c, &Xs[buf][r * XS_STRIDE + c]);
    }
    {
      int r = idx >> 6, c = idx & 63;
      copy16_g2l(Wh + (size_t)(k0 + r) * Fd + c, &Ws[buf][r * WS_STRIDE + c]);
    }
  };

  v8f acc0 = {}, acc1 = {};
  load_tiles(0, 0);

  for (int kt = 0; kt < INd / 32; ++kt) {
    const int buf = kt & 1;
    wait_async();
    __syncthreads();                     // tile[buf] complete; prev reads done
    if (kt + 1 < INd / 32) load_tiles((kt + 1) * 32, buf ^ 1);

    // A fragment: 16x32 f16 (ISA 7.12.2 layout)
    v16h afr;
#pragma unroll
    for (int vg = 0; vg < 8; ++vg) {
      int kb = ((vg & 4) << 2) + (khalf << 3) + ((vg & 3) << 1);
      afr[2 * vg]     = Xs[buf][(isub * 16 + mrow) * XS_STRIDE + kb];
      afr[2 * vg + 1] = Xs[buf][(isub * 16 + mrow) * XS_STRIDE + kb + 1];
    }
    // Two B fragments (32x16): lane%16 = N column, elements = K rows (table order)
    {
      v16h bfr;
#pragma unroll
      for (int hh = 0; hh < 16; ++hh)
        bfr[hh] = Ws[buf][(((hh & 8) << 1) + (khalf << 3) + (hh & 7)) * WS_STRIDE + fhalf * 32 + (lane & 15)];
      acc0 = __builtin_amdgcn_wmma_f32_16x16x32_f16(false, afr, false, bfr,
                                                    (short)0, acc0, false, false);
    }
    {
      v16h bfr;
#pragma unroll
      for (int hh = 0; hh < 16; ++hh)
        bfr[hh] = Ws[buf][(((hh & 8) << 1) + (khalf << 3) + (hh & 7)) * WS_STRIDE + fhalf * 32 + 16 + (lane & 15)];
      acc1 = __builtin_amdgcn_wmma_f32_16x16x32_f16(false, afr, false, bfr,
                                                    (short)0, acc1, false, false);
    }
  }

  // epilogue: C layout -> lane = N col, vgpr v -> row v + 8*(lane>>4)
#pragma unroll
  for (int half = 0; half < 2; ++half) {
    v8f acc = half ? acc1 : acc0;
    int fcol = fhalf * 32 + half * 16 + (lane & 15);
    float bias = vb[fcol];
#pragma unroll
    for (int vg = 0; vg < 8; ++vg) {
      int m = isub * 16 + vg + 8 * khalf;
      size_t grow = (size_t)(m0 + m);
      float val = acc[vg] + bias;
      Vf[grow * Fd + fcol] = val;
      Vh[grow * Fd + fcol] = (_Float16)val;
    }
  }
}

// ---------------------------------------------------------------------------
// Kernel 2: q[b,h,n] = V[b,n,:] . qW[:,h] + qb[h]  (and k).  32 rows per block.
// ---------------------------------------------------------------------------
__global__ __launch_bounds__(256) void k_qk(
    const float* __restrict__ Vf,
    const float* __restrict__ qW, const float* __restrict__ qb,
    const float* __restrict__ kW, const float* __restrict__ kb,
    float* __restrict__ qo, float* __restrict__ ko) {
  __shared__ float Vs[32 * 64];
  __shared__ float qWs[64 * 8];
  __shared__ float kWs[64 * 8];
  const int t  = threadIdx.x;
  const int r0 = blockIdx.x * 32;
  {
    int idx = t * 8;
    int r = idx >> 6, c = idx & 63;
    *(float4*)(Vs + idx)     = *(const float4*)(Vf + (size_t)(r0 + r) * Fd + c);
    *(float4*)(Vs + idx + 4) = *(const float4*)(Vf + (size_t)(r0 + r) * Fd + c + 4);
  }
  qWs[t] = qW[t]; qWs[t + 256] = qW[t + 256];
  kWs[t] = kW[t]; kWs[t + 256] = kW[t + 256];
  __syncthreads();

  const int r = t >> 3, h = t & 7;
  float sq = qb[h], sk = kb[h];
#pragma unroll 8
  for (int f = 0; f < Fd; ++f) {
    float v = Vs[r * 64 + f];
    sq += v * qWs[f * Hd + h];
    sk += v * kWs[f * Hd + h];
  }
  const int grow = r0 + r;
  const int b = grow >> 11, n = grow & (Nd - 1);
  qo[((size_t)b * Hd + h) * Nd + n] = sq;
  ko[((size_t)b * Hd + h) * Nd + n] = sk;
}

// ---------------------------------------------------------------------------
// Kernel 3: per (b,h) stats of k for closed-form softmax row max:
//   Mp = max_j leaky(k_j)                   (used when q_i >= 0)
//   Mn = min_j (k_j<=0 ? k_j : 0.01*k_j)    (used when q_i < 0)
// ---------------------------------------------------------------------------
__global__ __launch_bounds__(256) void k_stats(const float* __restrict__ ko,
                                               float* __restrict__ Mp,
                                               float* __restrict__ Mn) {
  __shared__ float smax[256];
  __shared__ float smin[256];
  const int bh = blockIdx.x, t = threadIdx.x;
  float mp = -3.402823466e38f, mn = 3.402823466e38f;
  for (int j = t; j < Nd; j += 256) {
    float kv = ko[(size_t)bh * Nd + j];
    float plus  = kv >= 0.f ? kv : NEG_SLOPE * kv;
    float minus = kv <= 0.f ? kv : NEG_SLOPE * kv;
    mp = fmaxf(mp, plus);
    mn = fminf(mn, minus);
  }
  smax[t] = mp; smin[t] = mn;
  __syncthreads();
  for (int s = 128; s > 0; s >>= 1) {
    if (t < s) {
      smax[t] = fmaxf(smax[t], smax[t + s]);
      smin[t] = fminf(smin[t], smin[t + s]);
    }
    __syncthreads();
  }
  if (t == 0) { Mp[bh] = smax[0]; Mn[bh] = smin[0]; }
}

// ---------------------------------------------------------------------------
// Kernel 4: fused softmax(leaky(q_i*k_j)) @ V via WMMA.
// Block = one (b,h), 32 i-rows. 8 waves = 2 i-subtiles x 4 f-subtiles.
// j-step 64: P tile 32x64 f16 = exp(leaky(q*k)-rmax); V tile 64x64 f16
// (double-buffered, async-LDS). 2 WMMAs/wave/iter, f32 accumulate.
// Row max is exact & precomputed, so no online rescaling is needed.
// ---------------------------------------------------------------------------
#define PS_STRIDE 72   // 64 + 8
#define VS_STRIDE 72   // 64 + 8

__global__ __launch_bounds__(256) void k_attn(
    const _Float16* __restrict__ Vh,
    const float* __restrict__ qo, const float* __restrict__ ko,
    const float* __restrict__ Mp, const float* __restrict__ Mn,
    float* __restrict__ out) {
  __shared__ _Float16 Ps[32 * PS_STRIDE];
  __shared__ _Float16 Vs[2][64 * VS_STRIDE];
  __shared__ float qs[32];
  __shared__ float rmax[32];
  __shared__ float rpart[256];
  __shared__ float rsum[32];

  const int t    = threadIdx.x;
  const int lane = t & 31;
  const int wv   = t >> 5;
  const int bh = blockIdx.x >> 6;      // 64 i-tiles per (b,h)
  const int it = blockIdx.x & 63;
  const int b = bh >> 3, h = bh & 7;
  const int i0 = it * 32;
  const int isub  = wv >> 2;           // 0..1
  const int fbase = (wv & 3) * 16;     // 0,16,32,48
  const int mrow  = lane & 15;
  const int khalf = lane >> 4;

  if (t < 32) {
    float qv = qo[(size_t)bh * Nd + i0 + t];
    qs[t]   = qv;
    rmax[t] = qv >= 0.f ? qv * Mp[bh] : qv * Mn[bh];
  }

  const _Float16* Vbase = Vh + (size_t)b * Nd * Fd;
  const float*    krow  = ko + (size_t)bh * Nd;

  auto load_v = [&](int jb, int buf) {   // 64x64 halves, 32B per thread
    int r = t >> 2, c = (t & 3) * 16;
    const _Float16* g = Vbase + (size_t)(jb + r) * Fd + c;
    _Float16* l = &Vs[buf][r * VS_STRIDE + c];
    copy16_g2l(g, l);
    copy16_g2l(g + 8, l + 8);
  };

  load_v(0, 0);
  __syncthreads();                      // qs/rmax visible

  const int   pr  = t >> 3;             // P row this thread fills (0..31)
  const int   pj0 = (t & 7) * 8;        // 8 j entries per thread
  const float qv  = qs[pr];
  const float rm  = rmax[pr];
  float psum = 0.f;

  v8f acc = {};

  for (int jt = 0; jt < Nd / 64; ++jt) {
    const int jb  = jt * 64;
    const int buf = jt & 1;

    wait_async();
    __syncthreads();                    // Vs[buf] ready; prev iter reads done
    if (jt + 1 < Nd / 64) load_v(jb + 64, buf ^ 1);

    // P tile: exp(leaky(q_i * k_j) - rmax_i)
#pragma unroll
    for (int u = 0; u < 8; ++u) {
      int j = pj0 + u;
      float kv = krow[jb + j];
      float s = qv * kv;
      s = s >= 0.f ? s : NEG_SLOPE * s;
      float p = __expf(s - rm);
      psum += p;
      Ps[pr * PS_STRIDE + j] = (_Float16)p;
    }
    __syncthreads();                    // P tile ready

    // Two K-halves: A fragments from P, B fragments from V
#pragma unroll
    for (int kh2 = 0; kh2 < 2; ++kh2) {
      v16h afr;
#pragma unroll
      for (int vg = 0; vg < 8; ++vg) {
        int kb = kh2 * 32 + ((vg & 4) << 2) + (khalf << 3) + ((vg & 3) << 1);
        afr[2 * vg]     = Ps[(isub * 16 + mrow) * PS_STRIDE + kb];
        afr[2 * vg + 1] = Ps[(isub * 16 + mrow) * PS_STRIDE + kb + 1];
      }
      v16h bfr;
#pragma unroll
      for (int hh = 0; hh < 16; ++hh)
        bfr[hh] = Vs[buf][(kh2 * 32 + ((hh & 8) << 1) + (khalf << 3) + (hh & 7)) * VS_STRIDE + fbase + (lane & 15)];
      acc = __builtin_amdgcn_wmma_f32_16x16x32_f16(false, afr, false, bfr,
                                                   (short)0, acc, false, false);
    }
  }

  // reduce row sums (8 partials per row)
  rpart[t] = psum;
  __syncthreads();
  if (t < 32) {
    float s = 0.f;
#pragma unroll
    for (int u = 0; u < 8; ++u) s += rpart[t * 8 + u];
    rsum[t] = s;
  }
  __syncthreads();

  // epilogue: out[b, i, h, f]
  const int fcol = fbase + (lane & 15);
#pragma unroll
  for (int vg = 0; vg < 8; ++vg) {
    int m = isub * 16 + vg + 8 * khalf;
    int i = i0 + m;
    float val = acc[vg] / rsum[m];
    out[(((size_t)b * Nd + i) * Hd + h) * Fd + fcol] = val;
  }
}

// ---------------------------------------------------------------------------
// Host launcher
// ---------------------------------------------------------------------------
extern "C" void kernel_launch(void* const* d_in, const int* in_sizes, int n_in,
                              void* d_out, int out_size, void* d_ws, size_t ws_size,
                              hipStream_t stream) {
  const float* X  = (const float*)d_in[0];
  const float* vW = (const float*)d_in[1];
  const float* vb = (const float*)d_in[2];
  const float* qW = (const float*)d_in[3];
  const float* qb = (const float*)d_in[4];
  const float* kW = (const float*)d_in[5];
  const float* kb = (const float*)d_in[6];
  float* out = (float*)d_out;

  char* ws = (char*)d_ws;
  size_t off = 0;
  auto carve = [&](size_t bytes) -> char* {
    char* p = ws + off;
    off = (off + bytes + 255) & ~(size_t)255;
    return p;
  };
  _Float16* Xh  = (_Float16*)carve((size_t)Bd * Nd * INd * 2);
  _Float16* Wh  = (_Float16*)carve((size_t)INd * Fd * 2);
  float*    Vf  = (float*)   carve((size_t)Bd * Nd * Fd * 4);
  _Float16* Vhh = (_Float16*)carve((size_t)Bd * Nd * Fd * 2);
  float*    qo  = (float*)   carve((size_t)Bd * Hd * Nd * 4);
  float*    ko  = (float*)   carve((size_t)Bd * Hd * Nd * 4);
  float*    MpP = (float*)   carve(256);
  float*    MnP = (float*)   carve(256);

  const int nX = Bd * Nd * INd;   // 2,097,152
  const int nW = INd * Fd;        // 16,384

  k_f32_to_f16<<<(nX + 255) / 256, 256, 0, stream>>>(X, Xh, nX);
  k_f32_to_f16<<<(nW + 255) / 256, 256, 0, stream>>>(vW, Wh, nW);

  k_gemm_v<<<(Bd * Nd) / 64, 256, 0, stream>>>(Xh, Wh, vb, Vf, Vhh);

  k_qk<<<(Bd * Nd) / 32, 256, 0, stream>>>(Vf, qW, qb, kW, kb, qo, ko);

  k_stats<<<Bd * Hd, 256, 0, stream>>>(ko, MpP, MnP);

  k_attn<<<Bd * Hd * (Nd / 32), 256, 0, stream>>>(Vhh, qo, ko, MpP, MnP, out);
}
